// MultiHeadAttention_38817914421285
// MI455X (gfx1250) — matrix operand order repaired and checked by reading, not compiled, not run on hardware
//
#include <hip/hip_runtime.h>

// ---------------------------------------------------------------------------
// MHA forward for MI455X (gfx1250, wave32, WMMA 16x16x32 f16 -> f32 accum)
// B=4, S=2048, E=1024, H=16, D=64
// GEMMs stage activations into LDS with the Tensor Data Mover (TDM),
// double-buffered, overlapped with WMMA compute.
// ---------------------------------------------------------------------------

typedef _Float16 h16;
typedef __attribute__((ext_vector_type(4)))  _Float16 v4h;
typedef __attribute__((ext_vector_type(8)))  _Float16 v8h;
typedef __attribute__((ext_vector_type(16))) _Float16 v16h;
typedef __attribute__((ext_vector_type(8)))  float    v8f;
typedef __attribute__((ext_vector_type(4)))  unsigned int u32x4;
typedef __attribute__((ext_vector_type(8)))  unsigned int u32x8;

#define EMBED 1024
#define NHEAD 16
#define HDIM  64
#define BATCH 4
#define SEQ   2048
#define MROWS (BATCH * SEQ)          // 8192
#define SP_STRIDE 2056               // padded half-stride for score rows
#define KCHUNK 256                   // halves of K staged per TDM transfer
#define NCHUNK (EMBED / KCHUNK)      // 4
#define ABUF_HALVES (16 * KCHUNK)    // 4096 halves = 8 KB per buffer

static __device__ __forceinline__ v8f wmma_f16(v16h a, v16h b, v8f c) {
    return __builtin_amdgcn_wmma_f32_16x16x32_f16(false, a, false, b,
                                                  (short)0, c, false, false);
}

// A-fragment (16x32 f16): per lane, row = lane&15; halves 0..15 at K-offset
// base + 16*(lane>=16) + 0..15 (contiguous).
static __device__ __forceinline__ v16h load_a_frag(const h16* p) {
    v8h lo = *(const v8h*)(p);
    v8h hi = *(const v8h*)(p + 8);
    return __builtin_shufflevector(lo, hi, 0, 1, 2, 3, 4, 5, 6, 7,
                                   8, 9, 10, 11, 12, 13, 14, 15);
}

// TDM: DMA a 16-row x KCHUNK-half f16 tile (row stride EMBED halves) from
// global memory into LDS at byte offset lds_off.  D# groups built in SGPRs;
// 2D tensor so groups 2/3 are omitted (NULL).  Tracked by TENSORcnt.
static __device__ __forceinline__ void tdm_load_tile(unsigned long long gaddr,
                                                     unsigned lds_off) {
    u32x4 g0;
    g0.x = 1u;                                        // count=1, user descriptor
    g0.y = lds_off;                                   // lds_addr (bytes)
    g0.z = (unsigned)gaddr;                           // global_addr[31:0]
    g0.w = ((unsigned)(gaddr >> 32) & 0x01FFFFFFu)    // global_addr[56:32]
         | (2u << 30);                                // type = 2 ("image")
    u32x8 g1;
    g1[0] = 0x00010000u;                              // data_size=1 -> 2 bytes
    g1[1] = (EMBED & 0xFFFFu) << 16;                  // tensor_dim0 lo16 (1024)
    g1[2] = (16u << 16) | (EMBED >> 16);              // tensor_dim1=16 | dim0 hi16
    g1[3] = ((unsigned)KCHUNK << 16);                 // tile_dim0 = 256
    g1[4] = 16u;                                      // tile_dim1 = 16, tile_dim2=0
    g1[5] = (unsigned)EMBED;                          // tensor_dim0_stride lo32
    g1[6] = 0u;
    g1[7] = 0u;
    asm volatile("tensor_load_to_lds %0, %1" :: "s"(g0), "s"(g1) : "memory");
}

// ---------------------------------------------------------------------------
// f32 -> f16 elementwise convert (x -> Xh), 4 elements per thread
// ---------------------------------------------------------------------------
__global__ __launch_bounds__(256) void k_convert_f16(const float* __restrict__ x,
                                                     h16* __restrict__ y, int n) {
    int i = (blockIdx.x * 256 + threadIdx.x) * 4;
    if (i < n) {
        float4 v = *(const float4*)(x + i);
        v4h o;
        o.x = (h16)v.x; o.y = (h16)v.y; o.z = (h16)v.z; o.w = (h16)v.w;
        *(v4h*)(y + i) = o;
    }
}

// ---------------------------------------------------------------------------
// Transpose 1024x1024 f32 weight -> N-major f16 (Wt[n][k] = W[k][n])
// ---------------------------------------------------------------------------
__global__ void k_transpose_w(const float* __restrict__ W, h16* __restrict__ Wt) {
    __shared__ float tile[32][33];
    int tx = threadIdx.x, ty = threadIdx.y;
    int k = blockIdx.y * 32 + ty;
    int n = blockIdx.x * 32 + tx;
    tile[ty][tx] = W[k * EMBED + n];
    __syncthreads();
    int n2 = blockIdx.x * 32 + ty;
    int k2 = blockIdx.y * 32 + tx;
    Wt[(size_t)n2 * EMBED + k2] = (h16)tile[tx][ty];
}

// ---------------------------------------------------------------------------
// GEMM: C[m][n] = sum_k A[m][k] * Wt[n][k] + bias[n]
//   A: [8192][1024] f16 row-major staged via TDM into LDS (shared by 8 waves)
//   Wt: [1024][1024] f16 N-major, streamed from global (L2-resident)
//   mode 0: store f16 as Q/K layout  [b][h][s][64]
//   mode 1: store f16 as V-transposed [b][h][64][s]  (packed 8-half store)
//   mode 2: store f32 row-major [m][1024]  (final output)
// One wave per 16x16 tile; 8 waves per block. grid = (512, 8)
// ---------------------------------------------------------------------------
__global__ __launch_bounds__(256) void k_gemm(const h16* __restrict__ A,
                                              const h16* __restrict__ Wt,
                                              const float* __restrict__ bias,
                                              h16* __restrict__ outH,
                                              float* __restrict__ outF,
                                              int mode) {
    extern __shared__ char lds[];
    h16* abuf = (h16*)lds;                       // 2 x 16 x KCHUNK halves

    const int tid  = threadIdx.x;
    const int lane = tid & 31;
    const int w    = tid >> 5;
    const int m0   = blockIdx.x * 16;
    const int n0   = (blockIdx.y * 8 + w) * 16;
    const int hi8  = (lane >> 4) << 3;   // 0 or 8
    const int hi16 = (lane >> 4) << 4;   // 0 or 16

    const unsigned long long abase =
        (unsigned long long)(const void*)A + (unsigned long long)m0 * EMBED * 2;
    const h16* brow = Wt + (size_t)(n0 + (lane & 15)) * EMBED + hi16;
    const h16* alds = abuf + (lane & 15) * KCHUNK + hi16;

    v8f acc = {};

    if (w == 0) tdm_load_tile(abase, 0);                   // prologue: chunk 0

    for (int c = 0; c < NCHUNK; ++c) {
        __syncthreads();              // previous chunk's compute done
        if (w == 0) {
            if (c + 1 < NCHUNK) {     // prefetch next chunk into other buffer
                tdm_load_tile(abase + (unsigned long long)(c + 1) * KCHUNK * 2,
                              ((c + 1) & 1) * (ABUF_HALVES * 2));
                __builtin_amdgcn_s_wait_tensorcnt(1);      // chunk c landed
            } else {
                __builtin_amdgcn_s_wait_tensorcnt(0);
            }
        }
        __syncthreads();              // chunk c visible to all waves

        const h16* ap = alds + (c & 1) * ABUF_HALVES;
        const h16* bp = brow + c * KCHUNK;
#pragma unroll
        for (int kk = 0; kk < KCHUNK / 32; ++kk) {
            v16h a = load_a_frag(ap + 32 * kk);            // ds_load_b128 x2
            v16h b = *(const v16h*)(bp + 32 * kk);         // global b128 x2
            acc = wmma_f16(a, b, acc);
        }
    }

    const int n  = n0 + (lane & 15);
    const float bv = bias[n];

    if (mode == 2) {
#pragma unroll
        for (int r = 0; r < 8; ++r) {
            int m = m0 + r + hi8;
            outF[(size_t)m * EMBED + n] = acc[r] + bv;
        }
    } else if (mode == 0) {   // Q / K : [b][h][s][64]
        const int h = n >> 6, d = n & 63;
#pragma unroll
        for (int r = 0; r < 8; ++r) {
            int m = m0 + r + hi8;
            int bi = m >> 11, s = m & 2047;
            outH[((size_t)(bi * NHEAD + h) * SEQ + s) * HDIM + d] = (h16)(acc[r] + bv);
        }
    } else {                  // V transposed : [b][h][64][s], s contiguous over r
        const int h = n >> 6, d = n & 63;
        const int bi = m0 >> 11;
        const int sbase = (m0 & 2047) + hi8;
        v8h pack;
#pragma unroll
        for (int r = 0; r < 8; ++r) pack[r] = (h16)(acc[r] + bv);
        *(v8h*)(outH + ((size_t)(bi * NHEAD + h) * HDIM + d) * SEQ + sbase) = pack;
    }
}

// ---------------------------------------------------------------------------
// Attention: per (q-tile of 16, head, batch). 8 waves, dynamic LDS.
//   Phase 1: scores = Q K^T * 1/8, masked, -> LDS f16 [16][SP_STRIDE]
//   Phase 2: row max / exp / row sum (normalization deferred to output)
//   Phase 3: O = P V  (split-K over wave pairs), / rowsum, -> AttnH f16 [m][1024]
// ---------------------------------------------------------------------------
__global__ __launch_bounds__(256) void k_attn(const h16* __restrict__ Q,
                                              const h16* __restrict__ K,
                                              const h16* __restrict__ Vt,
                                              const int* __restrict__ mask,
                                              h16* __restrict__ AttnH) {
    extern __shared__ char smem[];
    h16*   SP     = (h16*)smem;                               // 16 * 2056 halves
    float* red    = (float*)(smem + 16 * SP_STRIDE * 2);      // 256 floats
    float* rowmax = red + 256;                                // 16
    float* rowsum = rowmax + 16;                              // 16
    float* comb   = rowsum + 16;                              // 4*16*16 floats

    const int tid  = threadIdx.x;
    const int lane = tid & 31;
    const int w    = tid >> 5;
    const int hi8  = (lane >> 4) << 3;
    const int hi16 = (lane >> 4) << 4;

    const int s0 = blockIdx.x * 16;
    const int hh = blockIdx.y;
    const int bb = blockIdx.z;
    const size_t bh = (size_t)(bb * NHEAD + hh);

    // ---- Phase 1: scores ------------------------------------------------
    const h16* qrow = Q + (bh * SEQ + s0 + (lane & 15)) * HDIM;
    v16h a0 = load_a_frag(qrow + 0  + hi16);
    v16h a1 = load_a_frag(qrow + 32 + hi16);

#pragma unroll 2
    for (int kt = 0; kt < 16; ++kt) {
        const int kc = w * 256 + kt * 16;                 // k-column tile origin
        const h16* krow = K + (bh * SEQ + kc + (lane & 15)) * HDIM + hi16;
        v16h b0 = *(const v16h*)(krow);
        v16h b1 = *(const v16h*)(krow + 32);
        v8f c = {};
        c = wmma_f16(a0, b0, c);
        c = wmma_f16(a1, b1, c);
        const int mv = mask[bb * SEQ + kc + (lane & 15)];
#pragma unroll
        for (int r = 0; r < 8; ++r) {
            int q = r + hi8;
            float s = c[r] * 0.125f;                      // 1/sqrt(64)
            s = (mv == 0) ? -30000.0f : s;                // finite in f16
            SP[q * SP_STRIDE + kc + (lane & 15)] = (h16)s;
        }
    }
    __syncthreads();

    // ---- Phase 2: softmax stats (16 threads per row) --------------------
    const int row = tid >> 4;
    const int j   = tid & 15;
    float pmax = -3.0e38f;
    for (int t = 0; t < SEQ / 16; ++t)
        pmax = fmaxf(pmax, (float)SP[row * SP_STRIDE + j + 16 * t]);
    red[row * 16 + j] = pmax;
    __syncthreads();
    if (j == 0) {
        float m = red[row * 16];
        for (int i = 1; i < 16; ++i) m = fmaxf(m, red[row * 16 + i]);
        rowmax[row] = m;
    }
    __syncthreads();
    const float rm = rowmax[row];
    float psum = 0.0f;
    for (int t = 0; t < SEQ / 16; ++t) {
        int idx = row * SP_STRIDE + j + 16 * t;
        float e = __expf((float)SP[idx] - rm);
        SP[idx] = (h16)e;                                  // unnormalized P
        psum += e;
    }
    red[row * 16 + j] = psum;
    __syncthreads();
    if (j == 0) {
        float s = red[row * 16];
        for (int i = 1; i < 16; ++i) s += red[row * 16 + i];
        rowsum[row] = s;
    }
    __syncthreads();

    // ---- Phase 3: O = P V, split-K across wave pairs --------------------
    const int nt = w & 3, d0 = nt * 16;
    const int kh = w >> 2;                                 // k-half: 0 or 1
    const h16* vrow = Vt + (bh * HDIM + d0 + (lane & 15)) * SEQ + kh * 1024 + hi16;
    const h16* prow = SP + (lane & 15) * SP_STRIDE + kh * 1024 + hi16;

    v8f acc = {};
#pragma unroll 4
    for (int i = 0; i < 32; ++i) {
        v16h a = load_a_frag(prow + 32 * i);
        v16h b = *(const v16h*)(vrow + 32 * i);
        acc = wmma_f16(a, b, acc);
    }

    if (w >= 4) {
#pragma unroll
        for (int r = 0; r < 8; ++r)
            comb[(w - 4) * 256 + (r + hi8) * 16 + (lane & 15)] = acc[r];
    }
    __syncthreads();
    if (w < 4) {
#pragma unroll
        for (int r = 0; r < 8; ++r) {
            int q = r + hi8;
            float o = acc[r] + comb[w * 256 + q * 16 + (lane & 15)];
            o /= rowsum[q];
            AttnH[(size_t)(bb * SEQ + s0 + q) * EMBED + hh * HDIM + d0 + (lane & 15)]
                = (h16)o;
        }
    }
}

// ---------------------------------------------------------------------------
// Host-side orchestration
// ---------------------------------------------------------------------------
extern "C" void kernel_launch(void* const* d_in, const int* in_sizes, int n_in,
                              void* d_out, int out_size, void* d_ws, size_t ws_size,
                              hipStream_t stream) {
    (void)in_sizes; (void)n_in; (void)out_size; (void)ws_size;

    const float* x    = (const float*)d_in[0];
    const int*   mask = (const int*)d_in[1];
    const float* Wq   = (const float*)d_in[2];
    const float* bq   = (const float*)d_in[3];
    const float* Wk   = (const float*)d_in[4];
    const float* bk   = (const float*)d_in[5];
    const float* Wv   = (const float*)d_in[6];
    const float* bv   = (const float*)d_in[7];
    const float* Wo   = (const float*)d_in[8];
    const float* bo   = (const float*)d_in[9];
    float* out = (float*)d_out;

    // workspace carve-up (f16 buffers), total ~92.3 MB
    char* ws = (char*)d_ws;
    const size_t XH_B  = (size_t)MROWS * EMBED * 2;   // 16 MB
    const size_t WT_B  = (size_t)EMBED * EMBED * 2;   //  2 MB each
    h16* Xh    = (h16*)(ws);
    h16* Qb    = (h16*)(ws + XH_B);
    h16* Kb    = (h16*)(ws + 2 * XH_B);
    h16* Vtb   = (h16*)(ws + 3 * XH_B);
    h16* AttnH = (h16*)(ws + 4 * XH_B);
    h16* Wqt   = (h16*)(ws + 5 * XH_B);
    h16* Wkt   = (h16*)(ws + 5 * XH_B + WT_B);
    h16* Wvt   = (h16*)(ws + 5 * XH_B + 2 * WT_B);
    h16* Wot   = (h16*)(ws + 5 * XH_B + 3 * WT_B);

    // 1) x -> f16
    {
        int n = MROWS * EMBED;                         // 8388608
        k_convert_f16<<<n / 1024, 256, 0, stream>>>(x, Xh, n);
    }
    // 2) weights -> N-major f16
    {
        dim3 g(32, 32), b(32, 32);
        k_transpose_w<<<g, b, 0, stream>>>(Wq, Wqt);
        k_transpose_w<<<g, b, 0, stream>>>(Wk, Wkt);
        k_transpose_w<<<g, b, 0, stream>>>(Wv, Wvt);
        k_transpose_w<<<g, b, 0, stream>>>(Wo, Wot);
    }
    // 3) projections (TDM-staged A, double-buffered: 2 x 8 KB LDS)
    {
        dim3 g(MROWS / 16, EMBED / 16 / 8);            // (512, 8)
        size_t smem = 2 * ABUF_HALVES * 2;             // 16 KB
        k_gemm<<<g, 256, smem, stream>>>(Xh, Wqt, bq, Qb,  nullptr, 0);
        k_gemm<<<g, 256, smem, stream>>>(Xh, Wkt, bk, Kb,  nullptr, 0);
        k_gemm<<<g, 256, smem, stream>>>(Xh, Wvt, bv, Vtb, nullptr, 1);
    }
    // 4) attention
    {
        dim3 g(SEQ / 16, NHEAD, BATCH);                // (128, 16, 4)
        size_t smem = (size_t)16 * SP_STRIDE * 2       // scores/P (f16)
                    + 256 * 4                          // red
                    + 16 * 4 + 16 * 4                  // rowmax, rowsum
                    + 4 * 256 * 4;                     // comb
        k_attn<<<g, 256, smem, stream>>>(Qb, Kb, Vtb, mask, AttnH);
    }
    // 5) output projection (f32)
    {
        dim3 g(MROWS / 16, EMBED / 16 / 8);
        size_t smem = 2 * ABUF_HALVES * 2;
        k_gemm<<<g, 256, smem, stream>>>(AttnH, Wot, bo, nullptr, out, 2);
    }
}
